// TrueSplatAttentionLayer_49744311222299
// MI455X (gfx1250) — matrix operand order repaired and checked by reading, not compiled, not run on hardware
//
#include <hip/hip_runtime.h>
#include <hip/hip_bf16.h>
#include <stdint.h>

// ---------------------------------------------------------------------------
// TrueSplatAttention for MI455X (gfx1250, wave32, WMMA bf16 16x16x32)
// B=4, S=8192, D=1024, K=16
// Pipeline: prep -> zero -> convW -> affinity(+tokens->bf16) ->
//           GEMM(values) -> states(atomic WMMA) -> tout -> GEMM(out)
// ---------------------------------------------------------------------------

#define B_BATCH 4
#define S_SEQ   8192
#define D_DIM   1024
#define K_SPL   16

// LDS tile row stride: 40 bf16 = 20 dwords = 80 bytes.
// - rows 16B-aligned -> b128 LDS ops
// - 20*d mod 64 distinct over 16 lanes -> conflict-free fragment loads
#define LSTR 40

typedef __attribute__((ext_vector_type(16))) __bf16 v16bf;
typedef __attribute__((ext_vector_type(8)))  float  v8f;

union AFrag { v16bf v; uint32_t u[8]; };

__device__ __forceinline__ uint32_t pack2bf(float lo, float hi) {
  __bf16 a = (__bf16)lo, b = (__bf16)hi;   // backend: v_cvt_pk_bf16_f32
  uint16_t ua, ub;
  __builtin_memcpy(&ua, &a, 2);
  __builtin_memcpy(&ub, &b, 2);
  return (uint32_t)ua | ((uint32_t)ub << 16);
}
__device__ __forceinline__ uint16_t f2bf16(float f) {
  __bf16 a = (__bf16)f;
  uint16_t u; __builtin_memcpy(&u, &a, 2);
  return u;
}

// ---------------------------------------------------------------------------
// c_sq[k] = ||center_k||^2 ; sinv[k] = 0.5 / clip(exp(ls),0.1,2)^2
// ---------------------------------------------------------------------------
__global__ void k_prep(const float* __restrict__ centers,
                       const float* __restrict__ log_scales,
                       float* __restrict__ csq, float* __restrict__ sinv) {
  int n = threadIdx.x;
  if (n < K_SPL) {
    float s = 0.f;
    const float* c = centers + n * D_DIM;
    for (int d = 0; d < D_DIM; d += 4) {
      float4 v = *(const float4*)(c + d);
      s += v.x * v.x + v.y * v.y + v.z * v.z + v.w * v.w;
    }
    csq[n] = s;
    float sc = __expf(log_scales[n]);
    sc = fminf(fmaxf(sc, 0.1f), 2.0f);
    sinv[n] = 0.5f / (sc * sc);
  }
}

__global__ void k_zero(float* __restrict__ p, int n) {
  int i = blockIdx.x * blockDim.x + threadIdx.x;
  if (i < n) p[i] = 0.f;
}

// Convert both weight matrices fp32 -> bf16 (dword-packed pairs)
__global__ void k_convw(const float* __restrict__ wv, const float* __restrict__ wo,
                        uint32_t* __restrict__ wvb, uint32_t* __restrict__ wob,
                        int npairs) {
  int i = blockIdx.x * blockDim.x + threadIdx.x;
  if (i < npairs) {
    float2 a = ((const float2*)wv)[i];
    wvb[i] = pack2bf(a.x, a.y);
    float2 b = ((const float2*)wo)[i];
    wob[i] = pack2bf(b.x, b.y);
  }
}

// ---------------------------------------------------------------------------
// Affinities via WMMA (tokens x centers^T, N=16), fused with:
//   - per-token ||t||^2 (LDS ds_add_f32)
//   - single-pass tokens fp32 -> bf16 writeback (b128 in, b64 out)
// 8 waves x 16-row tiles = 128 rows/block; 256 blocks.
// ---------------------------------------------------------------------------
__global__ __launch_bounds__(256) void k_affinity(
    const float* __restrict__ tokens, const float* __restrict__ centers,
    const float* __restrict__ csq, const float* __restrict__ sinv,
    uint16_t* __restrict__ tokens_bf, float* __restrict__ aff) {
  __shared__ __bf16 sA[128 * LSTR];
  __shared__ float  sT[128];

  const int t    = threadIdx.x;
  const int w    = t >> 5;
  const int lane = t & 31;
  const int h    = lane >> 4;
  const int nn   = lane & 15;
  const long rowBase = (long)blockIdx.x * 128;

  for (int i = t; i < 128; i += 256) sT[i] = 0.f;
  __syncthreads();

  v8f acc = {};
  for (int k0 = 0; k0 < D_DIM; k0 += 32) {
    // 128x32 fp32 tile: float4 loads, bf16 pack, LDS + global writeback
#pragma unroll
    for (int i = 0; i < 4; ++i) {
      int c   = i * 256 + t;        // 1024 float4 chunks
      int row = c >> 3;
      int kq  = (c & 7) * 4;
      long g  = (rowBase + row) * D_DIM + k0 + kq;
      float4 f = *(const float4*)(tokens + g);
      uint2 pk = make_uint2(pack2bf(f.x, f.y), pack2bf(f.z, f.w));
      *(uint2*)&sA[row * LSTR + kq]  = pk;
      *(uint2*)(tokens_bf + g)       = pk;
      atomicAdd(&sT[row], f.x * f.x + f.y * f.y + f.z * f.z + f.w * f.w);
    }
    __syncthreads();

    // B fragment = centers^T 32x16 (fp32 -> bf16 on the fly); lane col = nn
    AFrag bfm;
#pragma unroll
    for (int v = 0; v < 8; ++v) {
      int kk = (v < 4) ? (8 * h + 2 * v) : (16 + 8 * h + 2 * (v - 4));
      float2 c2 = *(const float2*)(centers + (long)nn * D_DIM + k0 + kk);
      bfm.u[v] = pack2bf(c2.x, c2.y);
    }
    // A fragment for this wave's 16 rows; lane row = nn
    AFrag af;
    const uint32_t* r32 = (const uint32_t*)&sA[(w * 16 + nn) * LSTR];
#pragma unroll
    for (int v = 0; v < 4; ++v) af.u[v]     = r32[4 * h + v];
#pragma unroll
    for (int v = 0; v < 4; ++v) af.u[4 + v] = r32[8 + 4 * h + v];

    acc = __builtin_amdgcn_wmma_f32_16x16x32_bf16(false, af.v, false, bfm.v,
                                                  (short)0, acc, false, false);
    __syncthreads();
  }

  // dist^2 -> exp -> normalize across 16 splats (shuffle reduce within halves)
  float c2 = csq[nn];
  float si = sinv[nn];
#pragma unroll
  for (int v = 0; v < 8; ++v) {
    int row  = w * 16 + 8 * h + v;
    float d2 = fmaxf(sT[row] - 2.f * acc[v] + c2, 0.f);
    float a  = __expf(-d2 * si);
    float r  = a;
    r += __shfl_xor(r, 1, 32);
    r += __shfl_xor(r, 2, 32);
    r += __shfl_xor(r, 4, 32);
    r += __shfl_xor(r, 8, 32);
    aff[(rowBase + row) * K_SPL + nn] = a / (r + 1e-8f);
  }
}

// ---------------------------------------------------------------------------
// bf16 WMMA GEMM: C[MxN] = A[MxK] * B[KxN], A/B bf16 row-major.
// 128x128 block tile, BK=32, 8 waves (2x4), each wave 64x32 (4x2 WMMA tiles).
// Register-staged double buffering: one barrier per K-step, b128 global loads.
// ---------------------------------------------------------------------------
template <bool OUT_BF16>
__global__ __launch_bounds__(256) void k_gemm_bf16(
    const uint16_t* __restrict__ A, const uint16_t* __restrict__ Bm,
    void* __restrict__ Cout, int M, int N, int K) {
  __shared__ __bf16 sA[2][128 * LSTR];   // [row][k]
  __shared__ __bf16 sB[2][128 * LSTR];   // transposed: [n][k]

  const int t    = threadIdx.x;
  const int w    = t >> 5;
  const int lane = t & 31;
  const int h    = lane >> 4;
  const int nn   = lane & 15;
  const int wm   = w & 1;    // 2 waves along M (64 rows each)
  const int wn   = w >> 1;   // 4 waves along N (32 cols each)
  const long rowBase = (long)blockIdx.y * 128;
  const long colBase = (long)blockIdx.x * 128;

  uint4 ra[2], rb[2];

  auto loadA = [&](int k0) {
#pragma unroll
    for (int i = 0; i < 2; ++i) {
      int c = i * 256 + t;                       // 512 chunks of 8 bf16
      int row = c >> 2, kq = (c & 3) * 8;
      ra[i] = *(const uint4*)(A + (rowBase + row) * (long)K + k0 + kq);
    }
  };
  auto loadB = [&](int k0) {
#pragma unroll
    for (int i = 0; i < 2; ++i) {
      int c = i * 256 + t;
      int k = c >> 4, nq = (c & 15) * 8;
      rb[i] = *(const uint4*)(Bm + (long)(k0 + k) * N + colBase + nq);
    }
  };
  auto storeTiles = [&](int buf) {
#pragma unroll
    for (int i = 0; i < 2; ++i) {
      int c = i * 256 + t;
      int row = c >> 2, kq = (c & 3) * 8;
      *(uint4*)&sA[buf][row * LSTR + kq] = ra[i];   // b128 LDS store
    }
    uint16_t* pB = (uint16_t*)sB[buf];
#pragma unroll
    for (int i = 0; i < 2; ++i) {
      int c = i * 256 + t;
      int k = c >> 4, nq = (c & 15) * 8;
      uint32_t u[4] = {rb[i].x, rb[i].y, rb[i].z, rb[i].w};
#pragma unroll
      for (int j = 0; j < 4; ++j) {                 // transpose scatter
        pB[(nq + 2 * j + 0) * LSTR + k] = (uint16_t)(u[j] & 0xFFFFu);
        pB[(nq + 2 * j + 1) * LSTR + k] = (uint16_t)(u[j] >> 16);
      }
    }
  };

  v8f acc[4][2] = {};

  loadA(0); loadB(0);
  storeTiles(0);
  __syncthreads();

  const int steps = K / 32;
  for (int s = 0; s < steps; ++s) {
    const int cur = s & 1;
    if (s + 1 < steps) { loadA((s + 1) * 32); loadB((s + 1) * 32); }

    AFrag afr[4], bfr[2];
#pragma unroll
    for (int i = 0; i < 4; ++i) {
      const uint32_t* r32 =
          (const uint32_t*)&sA[cur][(wm * 64 + i * 16 + nn) * LSTR];
#pragma unroll
      for (int v = 0; v < 4; ++v) afr[i].u[v]     = r32[4 * h + v];
#pragma unroll
      for (int v = 0; v < 4; ++v) afr[i].u[4 + v] = r32[8 + 4 * h + v];
    }
#pragma unroll
    for (int j = 0; j < 2; ++j) {
      const uint32_t* r32 =
          (const uint32_t*)&sB[cur][(wn * 32 + j * 16 + nn) * LSTR];
#pragma unroll
      for (int v = 0; v < 8; ++v) bfr[j].u[v] = r32[(v < 4) ? (4 * h + v) : (8 + 4 * h + (v - 4))];
    }
#pragma unroll
    for (int i = 0; i < 4; ++i)
#pragma unroll
      for (int j = 0; j < 2; ++j)
        acc[i][j] = __builtin_amdgcn_wmma_f32_16x16x32_bf16(
            false, afr[i].v, false, bfr[j].v, (short)0, acc[i][j], false, false);

    if (s + 1 < steps) storeTiles(cur ^ 1);
    __syncthreads();
  }

  // Store C (tile layout: row = 8h+v, col = nn)
#pragma unroll
  for (int i = 0; i < 4; ++i) {
#pragma unroll
    for (int j = 0; j < 2; ++j) {
      long col = colBase + wn * 32 + j * 16 + nn;
#pragma unroll
      for (int v = 0; v < 8; ++v) {
        long row = rowBase + wm * 64 + i * 16 + 8 * h + v;
        if (OUT_BF16) {
          ((uint16_t*)Cout)[row * N + col] = f2bf16(acc[i][j][v]);
        } else {
          // final output, never re-read: keep it out of L2's working set
          __builtin_nontemporal_store(acc[i][j][v], (float*)Cout + row * N + col);
        }
      }
    }
  }
}

// ---------------------------------------------------------------------------
// states[b,k,d] += sum_s aff[b,s,k]*values[b,s,d]
// A = aff^T (16 splats x 32 s), B = values (32 s x 16 d). One wave per block,
// S split into 8 chunks, fp32 global atomic reduction.
// ---------------------------------------------------------------------------
__global__ __launch_bounds__(32) void k_states(
    const float* __restrict__ aff, const uint16_t* __restrict__ valuesb,
    float* __restrict__ states) {
  const int lane = threadIdx.x;
  const int h    = lane >> 4;
  const int nn   = lane & 15;
  const int d0   = blockIdx.x * 16;
  const int b    = blockIdx.z;
  const long tokBase = (long)b * S_SEQ + (long)blockIdx.y * (S_SEQ / 8);

  v8f acc = {};
  for (int step = 0; step < (S_SEQ / 8) / 32; ++step) {
    long s0 = tokBase + step * 32;
    AFrag af, bfm;
#pragma unroll
    for (int v = 0; v < 8; ++v) {   // A: row = splat (nn), k = s offset
      int kb = (v < 4) ? (8 * h + 2 * v) : (16 + 8 * h + 2 * (v - 4));
      float f0 = aff[(s0 + kb) * K_SPL + nn];
      float f1 = aff[(s0 + kb + 1) * K_SPL + nn];
      af.u[v] = pack2bf(f0, f1);
    }
#pragma unroll
    for (int v = 0; v < 8; ++v) {   // B: col = d (nn), k = s offset
      int kk = (v < 4) ? (8 * h + 2 * v) : (16 + 8 * h + 2 * (v - 4));
      uint32_t lo = valuesb[(s0 + kk)     * D_DIM + d0 + nn];
      uint32_t hi = valuesb[(s0 + kk + 1) * D_DIM + d0 + nn];
      bfm.u[v] = lo | (hi << 16);
    }
    acc = __builtin_amdgcn_wmma_f32_16x16x32_bf16(false, af.v, false, bfm.v,
                                                  (short)0, acc, false, false);
  }
#pragma unroll
  for (int v = 0; v < 8; ++v)
    atomicAdd(&states[((long)b * K_SPL + 8 * h + v) * D_DIM + d0 + nn], acc[v]);
}

// ---------------------------------------------------------------------------
// tout[b,s,d] = sum_k aff[b,s,k]*states[b,k,d]  (bf16 output)
// 16 tokens per block; states staged through LDS in two 16x512 fp32 halves.
// ---------------------------------------------------------------------------
__global__ __launch_bounds__(256) void k_tout(
    const float* __restrict__ aff, const float* __restrict__ states,
    uint16_t* __restrict__ toutb) {
  __shared__ float sS[K_SPL * 512];
  __shared__ float sAff[16 * K_SPL];

  const int t      = threadIdx.x;
  const long gbase = (long)blockIdx.x * 16;
  const int b      = (int)(gbase >> 13);   // /8192
  sAff[t] = aff[(gbase + (t >> 4)) * K_SPL + (t & 15)];
  const float* stb = states + (long)b * K_SPL * D_DIM;

  for (int half = 0; half < 2; ++half) {
    __syncthreads();
#pragma unroll
    for (int i = 0; i < 8; ++i) {        // 8192 floats as 2048 float4
      int p  = i * 256 + t;
      int k  = p >> 7;
      int d4 = (p & 127) * 4;
      *(float4*)&sS[k * 512 + d4] =
          *(const float4*)(stb + (long)k * D_DIM + half * 512 + d4);
    }
    __syncthreads();
    for (int tk = 0; tk < 16; ++tk) {
#pragma unroll
      for (int i = 0; i < 2; ++i) {
        int d = t + i * 256;
        float a = 0.f;
#pragma unroll
        for (int k = 0; k < K_SPL; ++k) a += sAff[tk * K_SPL + k] * sS[k * 512 + d];
        toutb[(gbase + tk) * D_DIM + half * 512 + d] = f2bf16(a);
      }
    }
  }
}

// ---------------------------------------------------------------------------
extern "C" void kernel_launch(void* const* d_in, const int* in_sizes, int n_in,
                              void* d_out, int out_size, void* d_ws, size_t ws_size,
                              hipStream_t stream) {
  (void)in_sizes; (void)n_in; (void)out_size; (void)ws_size;
  const float* tokens     = (const float*)d_in[0];
  const float* centers    = (const float*)d_in[1];
  const float* log_scales = (const float*)d_in[2];
  const float* W_v        = (const float*)d_in[3];
  const float* W_o        = (const float*)d_in[4];

  char* ws = (char*)d_ws;
  size_t off = 0;
  auto carve = [&](size_t bytes) -> void* {
    void* p = ws + off;
    off = (off + bytes + 255) & ~(size_t)255;
    return p;
  };
  const size_t BSD = (size_t)B_BATCH * S_SEQ * D_DIM;
  uint16_t* tokens_bf = (uint16_t*)carve(BSD * 2);
  uint16_t* values_bf = (uint16_t*)carve(BSD * 2);
  uint16_t* wv_bf     = (uint16_t*)carve((size_t)D_DIM * D_DIM * 2);
  uint16_t* wo_bf     = (uint16_t*)carve((size_t)D_DIM * D_DIM * 2);
  float*    aff       = (float*)carve((size_t)B_BATCH * S_SEQ * K_SPL * 4);
  float*    states    = (float*)carve((size_t)B_BATCH * K_SPL * D_DIM * 4);
  float*    csq       = (float*)carve(256);
  float*    sinv      = (float*)carve(256);
  uint16_t* tout_bf   = tokens_bf;  // tokens_bf dead after the values GEMM

  const int M = B_BATCH * S_SEQ;

  k_prep<<<1, 64, 0, stream>>>(centers, log_scales, csq, sinv);
  k_zero<<<(B_BATCH * K_SPL * D_DIM) / 256, 256, 0, stream>>>(
      states, B_BATCH * K_SPL * D_DIM);
  k_convw<<<(D_DIM * D_DIM / 2) / 256, 256, 0, stream>>>(
      W_v, W_o, (uint32_t*)wv_bf, (uint32_t*)wo_bf, D_DIM * D_DIM / 2);
  k_affinity<<<M / 128, 256, 0, stream>>>(tokens, centers, csq, sinv,
                                          tokens_bf, aff);
  dim3 gg(D_DIM / 128, M / 128);
  k_gemm_bf16<true><<<gg, 256, 0, stream>>>(tokens_bf, wv_bf, values_bf,
                                            M, D_DIM, D_DIM);
  k_states<<<dim3(D_DIM / 16, 8, B_BATCH), 32, 0, stream>>>(aff, values_bf,
                                                            states);
  k_tout<<<M / 16, 256, 0, stream>>>(aff, states, tout_bf);
  k_gemm_bf16<false><<<gg, 256, 0, stream>>>(tout_bf, wo_bf, (float*)d_out,
                                             M, D_DIM, D_DIM);
}
